// MyConv2D_5093831213628
// MI455X (gfx1250) — matrix operand order repaired and checked by reading, not compiled, not run on hardware
//
#include <hip/hip_runtime.h>

typedef __bf16 bf16;
typedef bf16  v16bf __attribute__((ext_vector_type(16)));
typedef float v8f   __attribute__((ext_vector_type(8)));

#define C_IN  128
#define C_OUT 256
#define HW    56

// ---------------------------------------------------------------------------
// Kernel 1: pack W (f32, OIHW) into bf16 WMMA A-fragment layout:
//   [pos(9)][cchunk(4)][octile(16)][lane(32)][half(16)]  (bf16)
// Half j of lane L maps to K (contiguous order, matching B reads):
//   K = j + (L>=16 ? 16:0) ; M = L&15
// 294912 bf16 = 576 KB -> permanently L2-resident (192 MB L2).
// ---------------------------------------------------------------------------
__global__ void conv_pack_weights(const float* __restrict__ W,
                                  bf16* __restrict__ wp) {
    int idx  = blockIdx.x * 256 + threadIdx.x;
    int j    = idx & 15;
    int lane = (idx >> 4) & 31;
    int ot   = (idx >> 9) & 15;
    int cc   = (idx >> 13) & 3;
    int pos  = idx >> 15;
    if (pos >= 9) return;
    int v = j >> 1, s = j & 1;
    int K = (v * 2 + s) + ((lane >= 16) ? 16 : 0);
    int oc = ot * 16 + (lane & 15);
    int c  = cc * 32 + K;
    int kh = pos / 3, kw = pos % 3;
    wp[idx] = (bf16)W[((oc * C_IN + c) * 3 + kh) * 3 + kw];
}

// ---------------------------------------------------------------------------
// Kernel 2: implicit-GEMM 3x3 conv, bf16 WMMA / f32 accumulate.
// Workgroup = 256 threads (8 waves). Tile: 128 oc x 4 output rows (4x56 px,
// padded to 4x64). Wave w: 16 oc x 256 px = 16 v8f accumulators.
// Per c-chunk (32 ch): stage input rows h0-1..h0+4 ONCE (shared by all kh
// taps of all 4 output rows), double-buffered in LDS so staging of chunk
// cc+1 overlaps compute of chunk cc.
// Inner: kh(3) x kw(3) x orow(4): preload 4 B frags, 4 back-to-back WMMAs
// -> 576 v_wmma per wave. Weight L2 traffic: 0.26 GB total (halved vs 2-row).
// ---------------------------------------------------------------------------
__launch_bounds__(256)
__global__ void conv3x3_wmma_bf16(const float* __restrict__ x,
                                  const bf16*  __restrict__ wp,
                                  const float* __restrict__ bias,
                                  float* __restrict__ out) {
    // [buf 2][row 6][col 66][c 32] bf16 = 50688 B
    __shared__ __align__(16) bf16 sX[2 * 6 * 66 * 32];

    const int tid  = threadIdx.x;
    const int lane = tid & 31;
    const int wave = tid >> 5;
    const int g    = lane >> 4;
    const int nl   = lane & 15;

    const int tile = blockIdx.x;             // 0..447
    const int n    = tile / 14;
    const int h0   = (tile % 14) * 4;        // first of 4 output rows
    const int ocb  = blockIdx.y * 128;
    const int ot   = blockIdx.y * 8 + wave;  // global 16-oc tile id

    v8f acc[16] = {};                        // [orow 4][t 4]

    const int c_ld = tid >> 3;               // 0..31
    const int jj   = tid & 7;

    auto stage = [&](int buf, int cc) {
        const float* xbase =
            x + ((long)n * C_IN + (cc * 32 + c_ld)) * (long)(HW * HW);
#pragma unroll
        for (int r = 0; r < 6; ++r) {
            const int  row = h0 + r - 1;               // h0-1 .. h0+4
            const bool rv  = (row >= 0) && (row < HW);
            const float* xr = xbase + (long)row * HW;
#pragma unroll
            for (int i = 0; i < 9; ++i) {
                int col = jj + i * 8;
                if (col < 66) {
                    int wg = col - 1;
                    float v = (rv && wg >= 0 && wg < HW) ? xr[wg] : 0.0f;
                    sX[((buf * 6 + r) * 66 + col) * 32 + c_ld] = (bf16)v;
                }
            }
        }
    };

    stage(0, 0);
    __syncthreads();

    int buf = 0;
    for (int cc = 0; cc < 4; ++cc) {
        if (cc < 3) {
            stage(buf ^ 1, cc + 1);          // overlap with compute below
            __builtin_prefetch(               // warm next chunk's weight tile
                (const void*)(wp + ((((cc + 1) * 16 + ot) * 32 + lane) * 16)),
                0, 0);
        }
#pragma unroll
        for (int kh = 0; kh < 3; ++kh) {
#pragma unroll
            for (int kw = 0; kw < 3; ++kw) {
                const int pos = kh * 3 + kw;
                const uint4* ap = (const uint4*)
                    (wp + ((((pos * 4 + cc) * 16 + ot) * 32 + lane) * 16));
                union { uint4 u[2]; v16bf v; } A;
                A.u[0] = ap[0]; A.u[1] = ap[1];
#pragma unroll
                for (int orow = 0; orow < 4; ++orow) {
                    const int rbase = ((buf * 6 + orow + kh) * 66) * 32;
                    union { uint4 u[2]; v16bf v; } B[4];
#pragma unroll
                    for (int t = 0; t < 4; ++t) {   // preload all 4 B frags
                        const int col = t * 16 + nl + kw;
                        const uint4* bp =
                            (const uint4*)&sX[rbase + col * 32 + g * 16];
                        B[t].u[0] = bp[0]; B[t].u[1] = bp[1];
                    }
#pragma unroll
                    for (int t = 0; t < 4; ++t)     // 4 back-to-back WMMAs
                        acc[orow * 4 + t] = __builtin_amdgcn_wmma_f32_16x16x32_bf16(
                            false, A.v, false, B[t].v, (short)0,
                            acc[orow * 4 + t], false, false);
                }
            }
        }
        __syncthreads();
        buf ^= 1;
    }

    // Epilogue: D layout -> oc_local = r + g*8, w_local = nl; add bias.
#pragma unroll
    for (int orow = 0; orow < 4; ++orow) {
#pragma unroll
        for (int t = 0; t < 4; ++t) {
            const int w = t * 16 + nl;
            if (w < HW) {
#pragma unroll
                for (int r = 0; r < 8; ++r) {
                    const int oc = ocb + wave * 16 + g * 8 + r;
                    out[(((long)n * C_OUT + oc) * HW + (h0 + orow)) * HW + w] =
                        acc[orow * 4 + t][r] + bias[oc];
                }
            }
        }
    }
}

extern "C" void kernel_launch(void* const* d_in, const int* in_sizes, int n_in,
                              void* d_out, int out_size, void* d_ws, size_t ws_size,
                              hipStream_t stream) {
    const float* x = (const float*)d_in[0];   // 32x128x56x56 f32
    const float* W = (const float*)d_in[1];   // 256x128x3x3 f32
    const float* b = (const float*)d_in[2];   // 256 f32
    float* out = (float*)d_out;
    bf16* wpack = (bf16*)d_ws;                // 294912 bf16 = 576 KB

    conv_pack_weights<<<1152, 256, 0, stream>>>(W, wpack);

    // 448 four-row tiles (32 n * 14), 2 oc-blocks of 128
    dim3 grid(448, 2);
    conv3x3_wmma_bf16<<<grid, 256, 0, stream>>>(x, wpack, b, out);
}
